// RNN_8220567404961
// MI455X (gfx1250) — matrix operand order repaired and checked by reading, not compiled, not run on hardware
//
#include <hip/hip_runtime.h>
#include <stdint.h>

#define SEQ 40
#define NC 128
#define TILE_B 64
#define THREADS 512   // 16 waves: 4 M-slices x 4 unit-quarters
#define XPAD 144      // padded bf16 row stride: 288B, 16B-aligned, conflict-reducing

typedef __attribute__((ext_vector_type(16))) __bf16 bf16x16;
typedef __attribute__((ext_vector_type(8)))  float  f32x8;

union Frag { bf16x16 v; uint4 q[2]; };
union Pack8 { unsigned short s[8]; uint4 q; };

__device__ __forceinline__ unsigned short f2bf(float f) {
  unsigned int u = __builtin_bit_cast(unsigned int, f);
  u += 0x7FFFu + ((u >> 16) & 1u);           // round-to-nearest-even
  return (unsigned short)(u >> 16);
}
__device__ __forceinline__ float sigm(float x) { return 1.0f / (1.0f + __expf(-x)); }
__device__ __forceinline__ float tanh_(float x) {
  float e = __expf(2.0f * x);
  return (e - 1.0f) / (e + 1.0f);
}
__device__ __forceinline__ f32x8 z8() {
  f32x8 v;
  #pragma unroll
  for (int r = 0; r < 8; ++r) v[r] = 0.0f;
  return v;
}

// ---- prep: swizzle Wk (256x512) / Wd (128x128) f32 -> bf16 B-fragment order.
// Tiles grouped by unit-quarter uq so each wave's tiles live in one 64KB LDS
// window reachable with DS immediate offsets.
// Wk tile_lin = ((uq*8 + kc)*4 + g)*2 + ug   (256 tiles, 512 elems each)
// Wd tile_lin = (uq*4 + kc)*2 + ug           (32 tiles)
// elem(tile, lane, j) = W[kc*32 + (j/8)*16 + (lane/16)*8 + j%8][ntile*16 + lane%16]
__global__ void prep_weights(const float* __restrict__ Wk, const float* __restrict__ Wd,
                             unsigned short* __restrict__ wk_sw,
                             unsigned short* __restrict__ wd_sw) {
  int idx = blockIdx.x * blockDim.x + threadIdx.x;
  if (idx < 131072) {
    int j = idx & 15, l = (idx >> 4) & 31, t = idx >> 9;
    int ug = t & 1, g = (t >> 1) & 3, kc = (t >> 3) & 7, uq = t >> 6;
    int ntile = g * 8 + uq * 2 + ug;          // z column tile (gate-major)
    int K = kc * 32 + ((j >> 3) << 4) + ((l >> 4) << 3) + (j & 7);
    int col = ntile * 16 + (l & 15);
    wk_sw[idx] = f2bf(Wk[K * 512 + col]);
  } else if (idx < 131072 + 16384) {
    int i2 = idx - 131072;
    int j = i2 & 15, l = (i2 >> 4) & 31, t = i2 >> 9;
    int ug = t & 1, kc = (t >> 1) & 3, uq = t >> 3;
    int ntile = uq * 2 + ug;
    int K = kc * 32 + ((j >> 3) << 4) + ((l >> 4) << 3) + (j & 7);
    int col = ntile * 16 + (l & 15);
    wd_sw[i2] = f2bf(Wd[K * 128 + col]);
  }
}

// ---- persistent LSTM: one workgroup owns 64 batch rows for all 40 steps ----
__global__ void __launch_bounds__(THREADS, 1)
lstm_persistent(const float* __restrict__ x,
                const float* __restrict__ bvec,
                const float* __restrict__ bd,
                const unsigned short* __restrict__ wk_sw,
                const unsigned short* __restrict__ wd_sw,
                float* __restrict__ out) {
  extern __shared__ unsigned short smem[];
  unsigned short* wk_lds = smem;                       // 131072 bf16 = 256KB
  unsigned short* x_lds  = smem + 131072;              // 64 x 144
  unsigned short* h_lds  = x_lds + TILE_B * XPAD;      // 64 x 144

  const int tid   = threadIdx.x;
  const int lane  = tid & 31;
  const int wave  = tid >> 5;
  const int m     = wave & 3;      // 16-row M slice
  const int uq    = wave >> 2;     // unit quarter (32 units)
  const int rbase = blockIdx.x * TILE_B;
  const int lcol  = lane & 15;
  const int lhi   = lane >> 4;

  // wave-constant bases: all B-fragment DS accesses become immediate offsets
  const unsigned short* wkb = wk_lds + (size_t)uq * 32768 + lane * 16;
  const unsigned short* wdb = wd_sw + (size_t)uq * 4096 + lane * 16;

  // stage swizzled Wk into LDS (contiguous 16B copies, resident all 40 steps)
  {
    const uint4* src = (const uint4*)wk_sw;
    uint4* dst = (uint4*)wk_lds;
    for (int i = tid; i < 16384; i += THREADS) dst[i] = src[i];
  }
  // h0 = 0
  {
    unsigned int* hz = (unsigned int*)h_lds;
    for (int i = tid; i < TILE_B * XPAD / 2; i += THREADS) hz[i] = 0u;
  }

  // per-lane biases (column = lane%16 within each gate tile)
  float bI[2], bJ[2], bF[2], bO[2], bD[2];
  #pragma unroll
  for (int ug = 0; ug < 2; ++ug) {
    int col = uq * 32 + ug * 16 + lcol;
    bI[ug] = bvec[0 * 128 + col];
    bJ[ug] = bvec[1 * 128 + col];
    bF[ug] = bvec[2 * 128 + col] + 1.0f;   // FORGET_BIAS
    bO[ug] = bvec[3 * 128 + col];
    bD[ug] = bd[col];
  }

  f32x8 c[2];                                // cell state, persistent in VGPRs
  c[0] = z8(); c[1] = z8();

  // x-tile loader mapping: each thread owns 16 floats of the 64x128 tile
  const int xr_row = tid >> 3;
  const int xr_cb  = (tid & 7) * 16;

  // preload x(t=0)
  {
    const float* xr = x + ((size_t)(rbase + xr_row) * SEQ + 0) * NC;
    #pragma unroll
    for (int i = 0; i < 2; ++i) {
      int cb = xr_cb + i * 8;
      float4 a = *(const float4*)(xr + cb);
      float4 b2 = *(const float4*)(xr + cb + 4);
      Pack8 p; p.s[0]=f2bf(a.x); p.s[1]=f2bf(a.y); p.s[2]=f2bf(a.z); p.s[3]=f2bf(a.w);
               p.s[4]=f2bf(b2.x);p.s[5]=f2bf(b2.y);p.s[6]=f2bf(b2.z);p.s[7]=f2bf(b2.w);
      *(uint4*)(x_lds + xr_row * XPAD + cb) = p.q;
    }
    __builtin_prefetch(xr + NC, 0, 0);       // t=1 rows
  }
  __syncthreads();

  #pragma unroll 1
  for (int t = 0; t < SEQ; ++t) {
    f32x8 hv[2];
    #pragma unroll 1
    for (int ug = 0; ug < 2; ++ug) {
      f32x8 acc[4];                          // i, j, f, o gate tiles (16x16 f32)
      #pragma unroll
      for (int g = 0; g < 4; ++g) acc[g] = z8();

      #pragma unroll
      for (int kc = 0; kc < 8; ++kc) {       // K = 256 (128 x-part + 128 h-part)
        const unsigned short* abase = (kc < 4)
            ? (x_lds + (m * 16 + lcol) * XPAD + kc * 32 + lhi * 8)
            : (h_lds + (m * 16 + lcol) * XPAD + (kc - 4) * 32 + lhi * 8);
        Frag A;
        A.q[0] = *(const uint4*)(abase);
        A.q[1] = *(const uint4*)(abase + 16);
        #pragma unroll
        for (int g = 0; g < 4; ++g) {
          const unsigned short* bb = wkb + (((kc * 4 + g) * 2) + ug) * 512;
          Frag B;
          B.q[0] = *(const uint4*)(bb);
          B.q[1] = *(const uint4*)(bb + 8);
          acc[g] = __builtin_amdgcn_wmma_f32_16x16x32_bf16(
              false, A.v, false, B.v, (short)0, acc[g], false, false);
        }
      }
      // LSTM cell update: pure register math on C-layout
      f32x8 cn = c[ug], hh;
      #pragma unroll
      for (int r = 0; r < 8; ++r) {
        float iv = sigm(acc[0][r] + bI[ug]);
        float jv = tanh_(acc[1][r] + bJ[ug]);
        float fv = sigm(acc[2][r] + bF[ug]);
        float ov = sigm(acc[3][r] + bO[ug]);
        float nc2 = cn[r] * fv + iv * jv;
        cn[r] = nc2;
        hh[r] = tanh_(nc2) * ov;
      }
      c[ug] = cn;
      hv[ug] = hh;
    }
    __syncthreads();                         // all waves done reading old h / x

    // write new h (bf16) per C-layout position
    #pragma unroll
    for (int ug = 0; ug < 2; ++ug) {
      #pragma unroll
      for (int r = 0; r < 8; ++r) {
        h_lds[(m * 16 + r + lhi * 8) * XPAD + uq * 32 + ug * 16 + lcol] = f2bf(hv[ug][r]);
      }
    }

    // rotate in x(t+1), prefetch t+2
    if (t + 1 < SEQ) {
      const float* xr = x + ((size_t)(rbase + xr_row) * SEQ + (t + 1)) * NC;
      #pragma unroll
      for (int i = 0; i < 2; ++i) {
        int cb = xr_cb + i * 8;
        float4 a = *(const float4*)(xr + cb);
        float4 b2 = *(const float4*)(xr + cb + 4);
        Pack8 p; p.s[0]=f2bf(a.x); p.s[1]=f2bf(a.y); p.s[2]=f2bf(a.z); p.s[3]=f2bf(a.w);
                 p.s[4]=f2bf(b2.x);p.s[5]=f2bf(b2.y);p.s[6]=f2bf(b2.z);p.s[7]=f2bf(b2.w);
        *(uint4*)(x_lds + xr_row * XPAD + cb) = p.q;
      }
      if (t + 2 < SEQ) __builtin_prefetch(xr + NC, 0, 0);
    }
    __syncthreads();
  }

  // final dense: logits = h @ Wd + bd  (K=128, Wd fragments from L2)
  #pragma unroll 1
  for (int ug = 0; ug < 2; ++ug) {
    f32x8 acc = z8();
    #pragma unroll
    for (int kc = 0; kc < 4; ++kc) {
      const unsigned short* abase = h_lds + (m * 16 + lcol) * XPAD + kc * 32 + lhi * 8;
      Frag A;
      A.q[0] = *(const uint4*)(abase);
      A.q[1] = *(const uint4*)(abase + 16);
      const unsigned short* bb = wdb + (kc * 2 + ug) * 512;
      Frag B;
      B.q[0] = *(const uint4*)(bb);
      B.q[1] = *(const uint4*)(bb + 8);
      acc = __builtin_amdgcn_wmma_f32_16x16x32_bf16(
          false, A.v, false, B.v, (short)0, acc, false, false);
    }
    #pragma unroll
    for (int r = 0; r < 8; ++r) {
      int row = rbase + m * 16 + r + lhi * 8;
      int col = uq * 32 + ug * 16 + lcol;
      out[(size_t)row * 128 + col] = acc[r] + bD[ug];
    }
  }
}

extern "C" void kernel_launch(void* const* d_in, const int* in_sizes, int n_in,
                              void* d_out, int out_size, void* d_ws, size_t ws_size,
                              hipStream_t stream) {
  const float* x  = (const float*)d_in[0];
  const float* Wk = (const float*)d_in[1];
  const float* b  = (const float*)d_in[2];
  const float* Wd = (const float*)d_in[3];
  const float* bd = (const float*)d_in[4];
  (void)in_sizes; (void)n_in; (void)out_size; (void)ws_size;
  unsigned short* wk_sw = (unsigned short*)d_ws;          // 131072 bf16
  unsigned short* wd_sw = wk_sw + 131072;                 // 16384 bf16

  prep_weights<<<(131072 + 16384) / 256, 256, 0, stream>>>(Wk, Wd, wk_sw, wd_sw);

  size_t smem = (size_t)(131072 + 2 * TILE_B * XPAD) * sizeof(unsigned short); // 299008 B
  (void)hipFuncSetAttribute((const void*)lstm_persistent,
                            hipFuncAttributeMaxDynamicSharedMemorySize, (int)smem);
  lstm_persistent<<<8192 / TILE_B, THREADS, smem, stream>>>(x, b, bd, wk_sw, wd_sw,
                                                            (float*)d_out);
}
